// TrackDetectionInteraction_3358664425670
// MI455X (gfx1250) — compile-verified, matrix operand order checked by reading, not run on hardware
//
#include <hip/hip_runtime.h>
#include <math.h>

// ---------------------------------------------------------------------------
// Problem constants (match the reference)
// ---------------------------------------------------------------------------
#define Bn   32
#define NT   512
#define ND   1024
#define Cc   256
#define Hh   4
#define HDd  64
#define NEG_INF_MASK (-1000000000.0f)

typedef __attribute__((ext_vector_type(8)))  float  v8f;
typedef __attribute__((ext_vector_type(16))) __bf16 v16bf;

// ---------------------------------------------------------------------------
// bf16 helpers (bit-level, RNE)
// ---------------------------------------------------------------------------
__device__ __forceinline__ unsigned short f32_to_bf16(float f) {
    unsigned int u = __float_as_uint(f);
    unsigned int r = u + 0x7FFFu + ((u >> 16) & 1u);
    return (unsigned short)(r >> 16);
}

union FragU {
    v16bf v;
    unsigned int u[8];
};

// WMMA A/B fragment loader for 16-bit data, per CDNA5 ISA §7.12.2:
//   lane L holds row/col (rowbase + L%16); VGPR p holds the k-pair at
//   k = k0 + 2*(p%4) + 16*(p>=4) + 8*(L>=16).
// Source must be contiguous along k (A row-major, B stored N-major).
__device__ __forceinline__ FragU load_frag_b16(const unsigned short* base,
                                               int stride, int rowbase, int k0) {
    FragU f;
    const int lane = threadIdx.x & 31;
    const int row  = rowbase + (lane & 15);
    const int hi8  = ((lane >> 4) & 1) << 3;
    const unsigned short* p = base + (size_t)row * stride + k0 + hi8;
#pragma unroll
    for (int i = 0; i < 8; ++i) {
        int k = 2 * (i & 3) + ((i >= 4) ? 16 : 0);
        f.u[i] = *(const unsigned int*)(p + k);
    }
    return f;
}

// Same fragment layout, but source is f32 (converted on load).
__device__ __forceinline__ FragU load_frag_f32(const float* base,
                                               int stride, int rowbase, int k0) {
    FragU f;
    const int lane = threadIdx.x & 31;
    const int row  = rowbase + (lane & 15);
    const int hi8  = ((lane >> 4) & 1) << 3;
    const float* p = base + (size_t)row * stride + k0 + hi8;
#pragma unroll
    for (int i = 0; i < 8; ++i) {
        int k = 2 * (i & 3) + ((i >= 4) ? 16 : 0);
        float2 d = *(const float2*)(p + k);
        f.u[i] = (unsigned int)f32_to_bf16(d.x) |
                 ((unsigned int)f32_to_bf16(d.y) << 16);
    }
    return f;
}

__device__ __forceinline__ v8f wmma_bf16(v16bf a, v16bf b, v8f c) {
    return __builtin_amdgcn_wmma_f32_16x16x32_bf16(false, a, false, b,
                                                   (short)0, c, false, false);
}

__device__ __forceinline__ v8f vzero8() {
    v8f z;
#pragma unroll
    for (int i = 0; i < 8; ++i) z[i] = 0.0f;
    return z;
}

// ---------------------------------------------------------------------------
// Kernel: convert f32 weight (K x N, row-major) -> bf16 transposed (N x K)
// ---------------------------------------------------------------------------
__global__ void convert_weight_kernel(const float* __restrict__ W,
                                      unsigned short* __restrict__ WT) {
    int idx = blockIdx.x * blockDim.x + threadIdx.x;   // 0 .. 65535
    int k = idx >> 8;
    int n = idx & 255;
    WT[(size_t)n * Cc + k] = f32_to_bf16(W[idx]);
}

// ---------------------------------------------------------------------------
// Generic WMMA GEMM:  out[M x 256] = A[M x 256] @ W[256 x 256] + bias
//   AMODE: 0 = A is f32, 1 = A is bf16
//   OMODE: 0 = bf16 out; 1 = relu -> bf16; 2 = bf16 transposed per batch;
//          3 = f32 out = resid + acc + bias
// Block: 256 threads (8 waves); tile 128(M) x 64(N); grid (M/128, 256/64)
// ---------------------------------------------------------------------------
template <int AMODE, int OMODE>
__global__ void gemm_kernel(const void* __restrict__ Ap,
                            const unsigned short* __restrict__ WT,
                            const float* __restrict__ bias,
                            const float* __restrict__ resid,
                            unsigned short* __restrict__ out_b16,
                            float* __restrict__ out_f32) {
    const int wave = threadIdx.x >> 5;
    const int lane = threadIdx.x & 31;
    const int m0 = blockIdx.x * 128 + wave * 16;
    const int n0 = blockIdx.y * 64;

    v8f acc[4];
#pragma unroll
    for (int t = 0; t < 4; ++t) acc[t] = vzero8();

    for (int k0 = 0; k0 < Cc; k0 += 32) {
        FragU a;
        if (AMODE == 0)
            a = load_frag_f32((const float*)Ap + (size_t)m0 * Cc, Cc, 0, k0);
        else
            a = load_frag_b16((const unsigned short*)Ap + (size_t)m0 * Cc, Cc, 0, k0);
#pragma unroll
        for (int t = 0; t < 4; ++t) {
            FragU b = load_frag_b16(WT, Cc, n0 + 16 * t, k0);
            acc[t] = wmma_bf16(a.v, b.v, acc[t]);
        }
    }

    const int hi8 = ((lane >> 4) & 1) << 3;
#pragma unroll
    for (int t = 0; t < 4; ++t) {
        const int n = n0 + 16 * t + (lane & 15);
        const float bval = bias[n];
#pragma unroll
        for (int r = 0; r < 8; ++r) {
            const int m = m0 + r + hi8;
            float val = acc[t][r] + bval;
            if (OMODE == 0) {
                out_b16[(size_t)m * Cc + n] = f32_to_bf16(val);
            } else if (OMODE == 1) {
                val = val > 0.0f ? val : 0.0f;
                out_b16[(size_t)m * Cc + n] = f32_to_bf16(val);
            } else if (OMODE == 2) {
                const int bb = m / ND, s = m % ND;          // transposed V store
                out_b16[((size_t)bb * Cc + n) * ND + s] = f32_to_bf16(val);
            } else {
                const size_t idx = (size_t)m * Cc + n;
                out_f32[idx] = resid[idx] + val;
            }
        }
    }
}

// ---------------------------------------------------------------------------
// Attention kernel: one block per (b, 16-row q tile); 4 waves = 1 wave/head.
// Flash-style online softmax over ND in chunks of 64; wave-parallel row
// stats (shfl_xor across the two 16-lane halves); next-chunk K/V prefetch
// (global_prefetch_b8); writes mean_logits + probs once, coalesced.
// ---------------------------------------------------------------------------
__global__ void attention_kernel(const unsigned short* __restrict__ q_b16,
                                 const unsigned short* __restrict__ k_b16,
                                 const unsigned short* __restrict__ vT_b16,
                                 const unsigned char* __restrict__ mask,
                                 unsigned short* __restrict__ attn_b16,
                                 float* __restrict__ mean_out,
                                 float* __restrict__ probs_out) {
    __shared__ float          S_s[Hh][16][64];
    __shared__ unsigned short P_s[Hh][16][64];
    __shared__ float          alpha_s[Hh][16];
    __shared__ float          l_s[Hh][16];

    const int b    = blockIdx.y;
    const int t0   = blockIdx.x * 16;
    const int h    = threadIdx.x >> 5;
    const int lane = threadIdx.x & 31;
    const int hi8  = ((lane >> 4) & 1) << 3;
    const float scale = 0.125f;                     // HD^-0.5

    const unsigned short* qb = q_b16 + ((size_t)b * NT + t0) * Cc + h * HDd;
    const unsigned short* kb = k_b16 + ((size_t)b * ND) * Cc + h * HDd;
    const unsigned short* vb = vT_b16 + ((size_t)b * Cc + h * HDd) * ND;

    FragU qf0 = load_frag_b16(qb, Cc, 0, 0);
    FragU qf1 = load_frag_b16(qb, Cc, 0, 32);

    v8f o[4];
#pragma unroll
    for (int t = 0; t < 4; ++t) o[t] = vzero8();

    float m_run = -INFINITY, l_run = 0.0f;

    for (int s0 = 0; s0 < ND; s0 += 64) {
        // --- prefetch next chunk of K and V (global_prefetch_b8) ---------
        if (s0 + 64 < ND) {
            const unsigned short* pk = kb + (size_t)(s0 + 64 + lane) * Cc;
            __builtin_prefetch(pk, 0, 1);
            __builtin_prefetch(pk + (size_t)32 * Cc, 0, 1);
            const unsigned short* pv = vb + (size_t)lane * ND + s0 + 64;
            __builtin_prefetch(pv, 0, 1);
            __builtin_prefetch(pv + (size_t)32 * ND, 0, 1);
        }

        // --- 1: S = (Q K^T) * scale, masked, into LDS --------------------
#pragma unroll
        for (int nt = 0; nt < 4; ++nt) {
            v8f sacc = vzero8();
            FragU kf0 = load_frag_b16(kb, Cc, s0 + 16 * nt, 0);
            sacc = wmma_bf16(qf0.v, kf0.v, sacc);
            FragU kf1 = load_frag_b16(kb, Cc, s0 + 16 * nt, 32);
            sacc = wmma_bf16(qf1.v, kf1.v, sacc);
            const int s = s0 + 16 * nt + (lane & 15);
            const bool masked = mask[(size_t)b * ND + s] != 0;
#pragma unroll
            for (int r = 0; r < 8; ++r) {
                float val = masked ? NEG_INF_MASK : sacc[r] * scale;
                S_s[h][r + hi8][16 * nt + (lane & 15)] = val;
            }
        }
        __syncthreads();

        // --- 2: mean over heads + sigmoid probs (block-wide) -------------
        for (int i = threadIdx.x; i < 16 * 64; i += 128) {
            const int row = i >> 6, col = i & 63;
            float sm = (S_s[0][row][col] + S_s[1][row][col] +
                        S_s[2][row][col] + S_s[3][row][col]) * 0.25f;
            const size_t oi = ((size_t)b * NT + t0 + row) * ND + s0 + col;
            mean_out[oi]  = sm;
            probs_out[oi] = 1.0f / (1.0f + __expf(-sm));
        }

        // --- 3: wave-parallel online-softmax stats fused with P write ----
        {
            const int row = lane & 15;
            const int cbase = (lane >> 4) * 32;     // each lane: half a row
            float cm = -INFINITY;
            for (int j = 0; j < 32; ++j)
                cm = fmaxf(cm, S_s[h][row][cbase + j]);
            cm = fmaxf(cm, __shfl_xor(cm, 16));     // combine halves (wave32)
            const float m_new = fmaxf(m_run, cm);
            const float al = __expf(m_run - m_new);
            float cs = 0.0f;
            for (int j = 0; j < 32; ++j) {
                float p = __expf(S_s[h][row][cbase + j] - m_new);
                cs += p;
                P_s[h][row][cbase + j] = f32_to_bf16(p);
            }
            cs += __shfl_xor(cs, 16);
            l_run = l_run * al + cs;
            m_run = m_new;
            if (lane < 16) alpha_s[h][row] = al;
        }
        __syncthreads();

        // --- 4: rescale accumulators; o += P @ V -------------------------
#pragma unroll
        for (int r = 0; r < 8; ++r) {
            const float al = alpha_s[h][r + hi8];
#pragma unroll
            for (int t = 0; t < 4; ++t) o[t][r] *= al;
        }
        FragU pf0 = load_frag_b16(&P_s[h][0][0], 64, 0, 0);
        FragU pf1 = load_frag_b16(&P_s[h][0][0], 64, 0, 32);
#pragma unroll
        for (int nt = 0; nt < 4; ++nt) {
            FragU vf0 = load_frag_b16(vb, ND, 16 * nt, s0);
            o[nt] = wmma_bf16(pf0.v, vf0.v, o[nt]);
            FragU vf1 = load_frag_b16(vb, ND, 16 * nt, s0 + 32);
            o[nt] = wmma_bf16(pf1.v, vf1.v, o[nt]);
        }
    }

    // --- finalize: out = o / l, write bf16 -------------------------------
    if (lane < 16) l_s[h][lane] = l_run;
    __syncthreads();
#pragma unroll
    for (int nt = 0; nt < 4; ++nt) {
#pragma unroll
        for (int r = 0; r < 8; ++r) {
            const int row = r + hi8;
            const float inv = 1.0f / l_s[h][row];
            const size_t idx = ((size_t)b * NT + t0 + row) * Cc + h * HDd +
                               16 * nt + (lane & 15);
            attn_b16[idx] = f32_to_bf16(o[nt][r] * inv);
        }
    }
}

// ---------------------------------------------------------------------------
// Fused output-projection + residual + LayerNorm.
// Block: 256 threads (8 waves), one 16-row tile, full N=256.
// LN reduction is a 2-level tree (sum + sum-of-squares).
// ---------------------------------------------------------------------------
__global__ void oproj_ln_kernel(const unsigned short* __restrict__ attn_b16,
                                const unsigned short* __restrict__ WoT,
                                const float* __restrict__ bo,
                                const float* __restrict__ track_query,
                                const float* __restrict__ ln_g,
                                const float* __restrict__ ln_b,
                                float* __restrict__ upd_f32,
                                unsigned short* __restrict__ upd_b16) {
    __shared__ float Y_s[16][Cc];
    __shared__ float psum[16][17], psq[16][17];     // padded: no bank conflict
    __shared__ float mu_s[16], rs_s[16];

    const int wave = threadIdx.x >> 5;
    const int lane = threadIdx.x & 31;
    const int m0 = blockIdx.x * 16;
    const int n0 = wave * 32;
    const int hi8 = ((lane >> 4) & 1) << 3;

    v8f acc[2];
    acc[0] = vzero8(); acc[1] = vzero8();
    for (int k0 = 0; k0 < Cc; k0 += 32) {
        FragU a = load_frag_b16(attn_b16 + (size_t)m0 * Cc, Cc, 0, k0);
#pragma unroll
        for (int t = 0; t < 2; ++t) {
            FragU bfr = load_frag_b16(WoT, Cc, n0 + 16 * t, k0);
            acc[t] = wmma_bf16(a.v, bfr.v, acc[t]);
        }
    }
#pragma unroll
    for (int t = 0; t < 2; ++t) {
        const int n = n0 + 16 * t + (lane & 15);
        const float bval = bo[n];
#pragma unroll
        for (int r = 0; r < 8; ++r) {
            const int mr = r + hi8;
            Y_s[mr][n] = acc[t][r] + bval +
                         track_query[(size_t)(m0 + mr) * Cc + n];
        }
    }
    __syncthreads();

    // level 1: each thread reduces a 16-wide segment of one row
    {
        const int row = threadIdx.x >> 4, seg = threadIdx.x & 15;
        float s = 0.0f, q = 0.0f;
#pragma unroll
        for (int j = 0; j < 16; ++j) {
            float x = Y_s[row][seg * 16 + j];
            s += x;
            q += x * x;
        }
        psum[row][seg] = s;
        psq[row][seg]  = q;
    }
    __syncthreads();

    // level 2: 16 threads combine per-row partials
    if (threadIdx.x < 16) {
        const int row = threadIdx.x;
        float ts = 0.0f, tq = 0.0f;
#pragma unroll
        for (int j = 0; j < 16; ++j) { ts += psum[row][j]; tq += psq[row][j]; }
        const float mu  = ts * (1.0f / Cc);
        float var = tq * (1.0f / Cc) - mu * mu;
        var = fmaxf(var, 0.0f);
        mu_s[row] = mu;
        rs_s[row] = rsqrtf(var + 1e-5f);
    }
    __syncthreads();

    for (int i = threadIdx.x; i < 16 * Cc; i += 256) {
        const int row = i >> 8, col = i & 255;
        float y = (Y_s[row][col] - mu_s[row]) * rs_s[row] * ln_g[col] + ln_b[col];
        const size_t idx = (size_t)(m0 + row) * Cc + col;
        upd_f32[idx] = y;
        upd_b16[idx] = f32_to_bf16(y);
    }
}

// ---------------------------------------------------------------------------
// Host-side launcher
// ---------------------------------------------------------------------------
extern "C" void kernel_launch(void* const* d_in, const int* in_sizes, int n_in,
                              void* d_out, int out_size, void* d_ws, size_t ws_size,
                              hipStream_t stream) {
    const float* track_query = (const float*)d_in[0];
    const float* context_det = (const float*)d_in[1];
    const unsigned char* mask = (const unsigned char*)d_in[2];  // jax bool -> u8
    const float* Wq = (const float*)d_in[3];
    const float* bq = (const float*)d_in[4];
    const float* Wk = (const float*)d_in[5];
    const float* bk = (const float*)d_in[6];
    const float* Wv = (const float*)d_in[7];
    const float* bv = (const float*)d_in[8];
    const float* Wo = (const float*)d_in[9];
    const float* bo = (const float*)d_in[10];
    const float* ln_g = (const float*)d_in[11];
    const float* ln_b = (const float*)d_in[12];
    const float* W1 = (const float*)d_in[13];
    const float* b1 = (const float*)d_in[14];
    const float* W2 = (const float*)d_in[15];
    const float* b2 = (const float*)d_in[16];
    (void)in_sizes; (void)n_in; (void)out_size; (void)ws_size;

    float* out = (float*)d_out;
    float* upd_out   = out;                                   // (B,NT,C)
    float* mean_out  = out + (size_t)Bn * NT * Cc;            // (B,NT,ND)
    float* probs_out = mean_out + (size_t)Bn * NT * ND;       // (B,NT,ND)

    // Workspace layout (256B-aligned sub-buffers)
    char* w = (char*)d_ws;
    size_t off = 0;
    auto take = [&](size_t bytes) { char* p = w + off; off = (off + bytes + 255) & ~(size_t)255; return p; };
    unsigned short* WqT = (unsigned short*)take(Cc * Cc * 2);
    unsigned short* WkT = (unsigned short*)take(Cc * Cc * 2);
    unsigned short* WvT = (unsigned short*)take(Cc * Cc * 2);
    unsigned short* WoT = (unsigned short*)take(Cc * Cc * 2);
    unsigned short* W1T = (unsigned short*)take(Cc * Cc * 2);
    unsigned short* W2T = (unsigned short*)take(Cc * Cc * 2);
    unsigned short* q_b16    = (unsigned short*)take((size_t)Bn * NT * Cc * 2);
    unsigned short* k_b16    = (unsigned short*)take((size_t)Bn * ND * Cc * 2);
    unsigned short* vT_b16   = (unsigned short*)take((size_t)Bn * Cc * ND * 2);
    unsigned short* attn_b16 = (unsigned short*)take((size_t)Bn * NT * Cc * 2);
    unsigned short* upd_b16  = (unsigned short*)take((size_t)Bn * NT * Cc * 2);
    unsigned short* h_b16    = (unsigned short*)take((size_t)Bn * NT * Cc * 2);
    float*          upd_f32  = (float*)take((size_t)Bn * NT * Cc * 4);

    // 1) weight convert + transpose to bf16 (N-major)
    convert_weight_kernel<<<256, 256, 0, stream>>>(Wq, WqT);
    convert_weight_kernel<<<256, 256, 0, stream>>>(Wk, WkT);
    convert_weight_kernel<<<256, 256, 0, stream>>>(Wv, WvT);
    convert_weight_kernel<<<256, 256, 0, stream>>>(Wo, WoT);
    convert_weight_kernel<<<256, 256, 0, stream>>>(W1, W1T);
    convert_weight_kernel<<<256, 256, 0, stream>>>(W2, W2T);

    // 2) Q/K/V projections (WMMA bf16)
    gemm_kernel<0, 0><<<dim3(Bn * NT / 128, 4), 256, 0, stream>>>(
        track_query, WqT, bq, nullptr, q_b16, nullptr);
    gemm_kernel<0, 0><<<dim3(Bn * ND / 128, 4), 256, 0, stream>>>(
        context_det, WkT, bk, nullptr, k_b16, nullptr);
    gemm_kernel<0, 2><<<dim3(Bn * ND / 128, 4), 256, 0, stream>>>(
        context_det, WvT, bv, nullptr, vT_b16, nullptr);

    // 3) attention (flash-style, writes mean_logits + probs on the fly)
    attention_kernel<<<dim3(NT / 16, Bn), 128, 0, stream>>>(
        q_b16, k_b16, vT_b16, mask, attn_b16, mean_out, probs_out);

    // 4) O-proj + residual + LayerNorm
    oproj_ln_kernel<<<Bn * NT / 16, 256, 0, stream>>>(
        attn_b16, WoT, bo, track_query, ln_g, ln_b, upd_f32, upd_b16);

    // 5) FFN: h = relu(upd @ W1 + b1);  upd_out = upd + h @ W2 + b2
    gemm_kernel<1, 1><<<dim3(Bn * NT / 128, 4), 256, 0, stream>>>(
        upd_b16, W1T, b1, nullptr, h_b16, nullptr);
    gemm_kernel<1, 3><<<dim3(Bn * NT / 128, 4), 256, 0, stream>>>(
        h_b16, W2T, b2, upd_f32, nullptr, upd_out);
}
